// HLUnit_13477607375168
// MI455X (gfx1250) — matrix operand order repaired and checked by reading, not compiled, not run on hardware
//
#include <hip/hip_runtime.h>

typedef __attribute__((ext_vector_type(16))) _Float16 v16h;
typedef __attribute__((ext_vector_type(8)))  _Float16 v8h;
typedef __attribute__((ext_vector_type(2)))  __fp16   v2h;   // cvt_pkrtz result type
typedef __attribute__((ext_vector_type(8)))  float    v8f;

#define NF 64
#define HO 254
#define WTILES 16            // ceil(254/16)
#define WAVES_PER_BLOCK 8
#define TILES_PER_WAVE 8
// total tiles = 12*254*16 = 48768 ; blocks = 48768/(8*8) = 762
#define NBLOCKS 762
#define ACT_STRIDE 72        // 16 pixels x 72 halves (144B rows, 16B aligned)

static __device__ __forceinline__ v8f wmma16(v16h a, v16h b, v8f c) {
  // v_wmma_f32_16x16x32_f16: D = A(16x32 f16) * B(32x16 f16) + C(16x16 f32)
  return __builtin_amdgcn_wmma_f32_16x16x32_f16(false, a, false, b, (short)0, c,
                                                false, false);
}
// First k-step: C = inline constant 0 (no accumulator registers needed)
static __device__ __forceinline__ v8f wmma16_z(v16h a, v16h b) {
  v8f z = {};
  return __builtin_amdgcn_wmma_f32_16x16x32_f16(false, a, false, b, (short)0, z,
                                                false, false);
}

// packed f16 ReLU: one v_pk_max_num_f16 (no canonicalize; WMMA yields no sNaN)
static __device__ __forceinline__ v2h relu_pk(v2h p) {
  v2h q;
  asm("v_pk_max_num_f16 %0, %1, 0" : "=v"(q) : "v"(p));
  return q;
}

// tanh(x) = (e^2x - 1) / (e^2x + 1); exp2/rcp hardware ops, TRANS co-executes
static __device__ __forceinline__ float fast_tanh(float x) {
  float e = __builtin_amdgcn_exp2f(2.8853900817779268f * x);  // e^(2x)
  float r = __builtin_amdgcn_rcpf(e + 1.0f);
  return (e - 1.0f) * r;
}

__global__ __launch_bounds__(256) void hl_mlp_pixelshuffle(
    const float* __restrict__ x,
    const float* __restrict__ w1, const float* __restrict__ b1,
    const float* __restrict__ w2, const float* __restrict__ b2,
    const float* __restrict__ w3, const float* __restrict__ b3,
    const float* __restrict__ w4, const float* __restrict__ b4,
    const float* __restrict__ w5, const float* __restrict__ b5,
    const float* __restrict__ w6, const float* __restrict__ b6,
    float* __restrict__ out)
{
  __shared__ __align__(16) _Float16 sW[4][NF][NF];                  // w2..w5 as f16
  __shared__ __align__(16) _Float16 sW6[16][NF];                    // w6 as f16
  __shared__ __align__(16) _Float16 sActs[WAVES_PER_BLOCK][16 * ACT_STRIDE];

  const int tid = threadIdx.x;

  // ---- stage weights (f32 -> f16) into LDS, cooperatively ----
  {
    _Float16* dst = &sW[0][0][0];
    for (int i = tid; i < NF * NF; i += 256) dst[0 * NF * NF + i] = (_Float16)w2[i];
    for (int i = tid; i < NF * NF; i += 256) dst[1 * NF * NF + i] = (_Float16)w3[i];
    for (int i = tid; i < NF * NF; i += 256) dst[2 * NF * NF + i] = (_Float16)w4[i];
    for (int i = tid; i < NF * NF; i += 256) dst[3 * NF * NF + i] = (_Float16)w5[i];
    _Float16* dst6 = &sW6[0][0];
    for (int i = tid; i < 16 * NF; i += 256) dst6[i] = (_Float16)w6[i];
  }
  __syncthreads();

  const int wid    = tid >> 5;
  const int lane   = tid & 31;
  const int nloc   = lane & 15;
  const bool hiH   = lane >= 16;
  const int koff   = hiH ? 8 : 0;   // A-frag K sub-block select per lane-half
  const int mhi    = hiH ? 8 : 0;   // C/D pixel-row offset per lane-half

  // ---- hoisted register fragments ----
  // Layer-1 B fragments: B[k][n] = w1[nt*16+n][k], k in {0,1,2}, rest zero.
  v16h B1f[4];
#pragma unroll
  for (int nt = 0; nt < 4; ++nt) {
    v16h f;
#pragma unroll
    for (int e = 0; e < 16; ++e) f[e] = (_Float16)0.0f;
    if (!hiH) {
      const float* wr = w1 + (nt * 16 + nloc) * 3;
      f[0] = (_Float16)wr[0];
      f[1] = (_Float16)wr[1];
      f[2] = (_Float16)wr[2];
    }
    B1f[nt] = f;
  }
  // Layer-6 B fragments: B[k][o] = w6[o][kt*32 + halfK + e]
  v16h B6f[2];
#pragma unroll
  for (int kt = 0; kt < 2; ++kt) {
    const float* wr = w6 + nloc * NF + kt * 32 + (hiH ? 16 : 0);
    v16h f;
#pragma unroll
    for (int e = 0; e < 16; ++e) f[e] = (_Float16)wr[e];
    B6f[kt] = f;
  }
  // Bias scalars (one per lane: feature = nloc within each n-tile)
  float bi1[4], bi2[4], bi3[4], bi4[4], bi5[4];
#pragma unroll
  for (int nt = 0; nt < 4; ++nt) {
    bi1[nt] = b1[nt * 16 + nloc];
    bi2[nt] = b2[nt * 16 + nloc];
    bi3[nt] = b3[nt * 16 + nloc];
    bi4[nt] = b4[nt * 16 + nloc];
    bi5[nt] = b5[nt * 16 + nloc];
  }
  const float bi6 = b6[nloc];

  _Float16* act = &sActs[wid][0];

  // A-fragment load: lane = pixel nloc; two contiguous 16B chunks per K-tile
  auto loadA = [&](int kt) -> v16h {
    const v8h* p0 = (const v8h*)(act + nloc * ACT_STRIDE + kt * 32 + 2 * koff);
    const v8h* p1 = (const v8h*)(act + nloc * ACT_STRIDE + kt * 32 + 8 + 2 * koff);
    v8h lo = *p0;
    v8h hi = *p1;
    return __builtin_shufflevector(lo, hi, 0, 1, 2, 3, 4, 5, 6, 7,
                                   8, 9, 10, 11, 12, 13, 14, 15);
  };
  // B-fragment load from staged weights: lane = out-feature, 16 contiguous K halves
  auto loadB = [&](const _Float16* wb, int nt, int kt) -> v16h {
    const v8h* p =
        (const v8h*)(wb + (nt * 16 + nloc) * NF + kt * 32 + (hiH ? 16 : 0));
    v8h lo = p[0];
    v8h hi = p[1];
    return __builtin_shufflevector(lo, hi, 0, 1, 2, 3, 4, 5, 6, 7,
                                   8, 9, 10, 11, 12, 13, 14, 15);
  };
  // D -> act transpose through LDS: lane holds feature f, VGPR r holds pixel.
  // Bias add (f32, dual-issue) -> packed cvt (v_cvt_pk_f16_f32, 2 elems/op)
  // -> packed ReLU (v_pk_max_num_f16) -> strided 16-bit DS stores.
  auto storeAct = [&](int nt, v8f c, float b) {
    _Float16* p0 = act + mhi * ACT_STRIDE + nt * 16 + nloc;
#pragma unroll
    for (int r = 0; r < 8; r += 2) {
      v2h p = __builtin_amdgcn_cvt_pkrtz(c[r] + b, c[r + 1] + b);
      v2h q = relu_pk(p);
      p0[r * ACT_STRIDE]       = q[0];
      p0[(r + 1) * ACT_STRIDE] = q[1];
    }
  };

#define MLP_LAYER(LI, BI)                                            \
  {                                                                  \
    const _Float16* wb = &sW[LI][0][0];                              \
    v16h A0 = loadA(0);                                              \
    v16h A1 = loadA(1);                                              \
    v8f c0 = wmma16_z(A0, loadB(wb, 0, 0));                          \
    c0 = wmma16(A1, loadB(wb, 0, 1), c0);                            \
    v8f c1 = wmma16_z(A0, loadB(wb, 1, 0));                          \
    c1 = wmma16(A1, loadB(wb, 1, 1), c1);                            \
    v8f c2 = wmma16_z(A0, loadB(wb, 2, 0));                          \
    c2 = wmma16(A1, loadB(wb, 2, 1), c2);                            \
    v8f c3 = wmma16_z(A0, loadB(wb, 3, 0));                          \
    c3 = wmma16(A1, loadB(wb, 3, 1), c3);                            \
    storeAct(0, c0, BI[0]);                                          \
    storeAct(1, c1, BI[1]);                                          \
    storeAct(2, c2, BI[2]);                                          \
    storeAct(3, c3, BI[3]);                                          \
  }

  // Layer-1 A fragment lives across the tile loop: only the 3 tap elements
  // (lanes 0-15, K=0..2) change per tile; the zero padding persists.
  v16h A1frag;
#pragma unroll
  for (int e = 0; e < 16; ++e) A1frag[e] = (_Float16)0.0f;

  const int gw = blockIdx.x * WAVES_PER_BLOCK + wid;
  for (int t = 0; t < TILES_PER_WAVE; ++t) {
    const int tile  = gw * TILES_PER_WAVE + t;
    const int wt    = tile & (WTILES - 1);
    const int rest  = tile >> 4;
    const int h     = rest % HO;
    const int n     = rest / HO;
    const int wbase = wt * 16;

    // ---- layer 1: 3 horizontal taps -> 64 features (K zero-padded to 32) ----
    {
      if (!hiH) {
        int wp = wbase + nloc;
        if (wp > HO - 1) wp = HO - 1;  // tail clamp (stores masked later)
        const float* px = x + ((size_t)n * 256 + h) * 256 + wp;
        A1frag[0] = (_Float16)px[0];   // v0 = x[h][w]
        A1frag[1] = (_Float16)px[1];   // v1 = x[h][w+1]
        A1frag[2] = (_Float16)px[2];   // v2 = x[h][w+2]   (KTYPE 'h')
      }
      v8f c0 = wmma16_z(A1frag, B1f[0]);
      v8f c1 = wmma16_z(A1frag, B1f[1]);
      v8f c2 = wmma16_z(A1frag, B1f[2]);
      v8f c3 = wmma16_z(A1frag, B1f[3]);
      storeAct(0, c0, bi1[0]);
      storeAct(1, c1, bi1[1]);
      storeAct(2, c2, bi1[2]);
      storeAct(3, c3, bi1[3]);
    }

    // ---- layers 2..5: 64 -> 64 ----
    MLP_LAYER(0, bi2)
    MLP_LAYER(1, bi3)
    MLP_LAYER(2, bi4)
    MLP_LAYER(3, bi5)

    // ---- layer 6: 64 -> 16, tanh, pixel-shuffle store ----
    {
      v16h A0 = loadA(0);
      v16h A1 = loadA(1);
      v8f c = wmma16_z(A0, B6f[0]);
      c = wmma16(A1, B6f[1], c);

      const int o     = nloc;                 // output sub-pixel channel
      const int orow  = h * 4 + (o >> 2);     // u1 = o/4
      const int ocol  = o & 3;                // u2 = o%4
      float* ob = out + (size_t)n * 1016 * 1016 + (size_t)orow * 1016 + ocol;
      if (wbase + mhi + 7 < HO) {
        // full-width tile: unmasked stores
#pragma unroll
        for (int r = 0; r < 8; ++r)
          ob[(wbase + mhi + r) * 4] = fast_tanh(c[r] + bi6);
      } else {
#pragma unroll
        for (int r = 0; r < 8; ++r) {
          int wp = wbase + mhi + r;
          if (wp < HO) ob[wp * 4] = fast_tanh(c[r] + bi6);
        }
      }
    }
  }
#undef MLP_LAYER
}

extern "C" void kernel_launch(void* const* d_in, const int* in_sizes, int n_in,
                              void* d_out, int out_size, void* d_ws, size_t ws_size,
                              hipStream_t stream) {
  (void)in_sizes; (void)n_in; (void)d_ws; (void)ws_size; (void)out_size;
  const float* x  = (const float*)d_in[0];
  const float* w1 = (const float*)d_in[1];
  const float* b1 = (const float*)d_in[2];
  const float* w2 = (const float*)d_in[3];
  const float* b2 = (const float*)d_in[4];
  const float* w3 = (const float*)d_in[5];
  const float* b3 = (const float*)d_in[6];
  const float* w4 = (const float*)d_in[7];
  const float* b4 = (const float*)d_in[8];
  const float* w5 = (const float*)d_in[9];
  const float* b5 = (const float*)d_in[10];
  const float* w6 = (const float*)d_in[11];
  const float* b6 = (const float*)d_in[12];
  float* out = (float*)d_out;

  hl_mlp_pixelshuffle<<<NBLOCKS, 256, 0, stream>>>(
      x, w1, b1, w2, b2, w3, b3, w4, b4, w5, b5, w6, b6, out);
}
